// MoE_51230369907077
// MI455X (gfx1250) — matrix operand order repaired and checked by reading, not compiled, not run on hardware
//
#include <hip/hip_runtime.h>
#include <cmath>

#define NTOK 8192
#define DDIM 1024
#define FDIM 4096
#define NEXP 8

typedef __attribute__((ext_vector_type(16))) __bf16    v16bf;
typedef __attribute__((ext_vector_type(8)))  float     v8f;
typedef __attribute__((ext_vector_type(4)))  unsigned  u32x4;
typedef __attribute__((ext_vector_type(8)))  unsigned  u32x8;

// ---------------------------------------------------------------------------
// WMMA fragment loaders (wave32, 16x16x32 bf16), LDS-resident tiles.
// A tile: [64][40] bf16, rows = M, cols = K. ISA layout:
//   lanes 0-15: M=lane, K in {0..7, 16..23}; lanes 16-31: M=lane-16, K in {8..15, 24..31}
// B tile stored transposed: [128][40] bf16, rows = N, cols = K so frag reads
// are two aligned 16-byte LDS loads per lane.
// ---------------------------------------------------------------------------
__device__ __forceinline__ v16bf load_a_frag(const __bf16 (*As)[40], int lane, int mBase) {
  int m  = mBase + (lane & 15);
  int kb = (lane >> 4) * 8;
  union { uint4 u[2]; v16bf v; } f;
  f.u[0] = *(const uint4*)(&As[m][kb]);
  f.u[1] = *(const uint4*)(&As[m][kb + 16]);
  return f.v;
}

__device__ __forceinline__ v16bf load_b_frag(const __bf16 (*Bs)[40], int lane, int nBase) {
  int n  = nBase + (lane & 15);
  int kb = (lane >> 4) * 8;
  union { uint4 u[2]; v16bf v; } f;
  f.u[0] = *(const uint4*)(&Bs[n][kb]);
  f.u[1] = *(const uint4*)(&Bs[n][kb + 16]);
  return f.v;
}

// Stage a 32(K) x 128(N) fp32 weight tile into transposed bf16 LDS [n][k],
// packing row-pairs so stores are b32.  256 threads.
__device__ __forceinline__ void stage_b_tile(__bf16 (*Bs)[40], const float* __restrict__ w,
                                             int ldw, int k0, int n0, int tid) {
  #pragma unroll
  for (int it = 0; it < 2; ++it) {
    const int linear = tid * 4 + it * 1024;        // 2048 = 16 row-pairs * 128 cols
    const int rp  = linear >> 7;                   // 0..15
    const int col = linear & 127;
    const float4 flo = *(const float4*)(w + (size_t)(k0 + 2 * rp)     * ldw + n0 + col);
    const float4 fhi = *(const float4*)(w + (size_t)(k0 + 2 * rp + 1) * ldw + n0 + col);
    union { __bf16 h[2]; unsigned u; } p;
    p.h[0] = (__bf16)flo.x; p.h[1] = (__bf16)fhi.x; *(unsigned*)&Bs[col + 0][2 * rp] = p.u;
    p.h[0] = (__bf16)flo.y; p.h[1] = (__bf16)fhi.y; *(unsigned*)&Bs[col + 1][2 * rp] = p.u;
    p.h[0] = (__bf16)flo.z; p.h[1] = (__bf16)fhi.z; *(unsigned*)&Bs[col + 2][2 * rp] = p.u;
    p.h[0] = (__bf16)flo.w; p.h[1] = (__bf16)fhi.w; *(unsigned*)&Bs[col + 3][2 * rp] = p.u;
  }
}

// ---------------------------------------------------------------------------
// Kernel 0: zero output (atomically accumulated later) and counters.
// ---------------------------------------------------------------------------
__global__ __launch_bounds__(256) void moe_zero_kernel(float* __restrict__ out,
                                                       float* __restrict__ impSum,
                                                       int*   __restrict__ cnt) {
  size_t i = (size_t)blockIdx.x * blockDim.x + threadIdx.x;   // covers N*D/4
  float4 z = make_float4(0.f, 0.f, 0.f, 0.f);
  ((float4*)out)[i] = z;
  if (blockIdx.x == 0 && threadIdx.x < NEXP) {
    impSum[threadIdx.x] = 0.f;
    cnt[threadIdx.x]    = 0;
  }
}

// ---------------------------------------------------------------------------
// Kernel 1: router. One wave32 per token.
// ---------------------------------------------------------------------------
__global__ __launch_bounds__(256) void moe_router_kernel(const float* __restrict__ X,
                                                         const float* __restrict__ rw,
                                                         const float* __restrict__ rb,
                                                         float* __restrict__ impSum,
                                                         int*   __restrict__ cnt,
                                                         int*   __restrict__ tokIdx,
                                                         float* __restrict__ gateVal,
                                                         __bf16* __restrict__ Xbf) {
  const int lane = threadIdx.x & 31;
  const int n    = blockIdx.x * 8 + (threadIdx.x >> 5);
  const float* xr = X + (size_t)n * DDIM;

  float acc[NEXP] = {0.f, 0.f, 0.f, 0.f, 0.f, 0.f, 0.f, 0.f};
  #pragma unroll 4
  for (int j = 0; j < DDIM / 32; ++j) {
    const int d = lane + j * 32;
    const float xv = xr[d];
    Xbf[(size_t)n * DDIM + d] = (__bf16)xv;               // fused fp32->bf16 activation cast
    const float4 wlo = *(const float4*)(rw + (size_t)d * NEXP);
    const float4 whi = *(const float4*)(rw + (size_t)d * NEXP + 4);
    acc[0] += xv * wlo.x; acc[1] += xv * wlo.y; acc[2] += xv * wlo.z; acc[3] += xv * wlo.w;
    acc[4] += xv * whi.x; acc[5] += xv * whi.y; acc[6] += xv * whi.z; acc[7] += xv * whi.w;
  }
  #pragma unroll
  for (int e = 0; e < NEXP; ++e) {
    float v = acc[e];
    #pragma unroll
    for (int off = 16; off > 0; off >>= 1) v += __shfl_xor(v, off, 32);
    acc[e] = v;
  }

  if (lane == 0) {
    float p[NEXP];
    float mx = -1e30f;
    #pragma unroll
    for (int e = 0; e < NEXP; ++e) { p[e] = acc[e] + rb[e]; mx = fmaxf(mx, p[e]); }
    float s = 0.f;
    #pragma unroll
    for (int e = 0; e < NEXP; ++e) { p[e] = __expf(p[e] - mx); s += p[e]; }
    const float inv = 1.f / s;
    #pragma unroll
    for (int e = 0; e < NEXP; ++e) { p[e] *= inv; atomicAdd(&impSum[e], p[e]); }
    int i1 = 0;
    #pragma unroll
    for (int e = 1; e < NEXP; ++e) if (p[e] > p[i1]) i1 = e;
    int i2 = (i1 == 0) ? 1 : 0;
    #pragma unroll
    for (int e = 0; e < NEXP; ++e) if (e != i1 && p[e] > p[i2]) i2 = e;
    const float norm = p[i1] + p[i2];
    int pos = atomicAdd(&cnt[i1], 1);
    tokIdx[i1 * NTOK + pos]  = n;
    gateVal[i1 * NTOK + pos] = p[i1] / norm;
    pos = atomicAdd(&cnt[i2], 1);
    tokIdx[i2 * NTOK + pos]  = n;
    gateVal[i2 * NTOK + pos] = p[i2] / norm;
  }
}

// ---------------------------------------------------------------------------
// Kernel 2: prefix-sum row offsets + aux loss.
// ---------------------------------------------------------------------------
__global__ void moe_prep_kernel(const float* __restrict__ impSum,
                                const int*   __restrict__ cnt,
                                int* __restrict__ rowStart,
                                float* __restrict__ outAux) {
  if (threadIdx.x == 0) {
    int acc = 0;
    #pragma unroll
    for (int e = 0; e < NEXP; ++e) { rowStart[e] = acc; acc += cnt[e]; }
    float aux = 0.f;
    #pragma unroll
    for (int e = 0; e < NEXP; ++e) {
      const float d = impSum[e] * (1.f / (float)NTOK) - (1.f / (float)NEXP);
      aux += d * d;
    }
    outAux[0] = aux * (1.f / (float)NEXP);
  }
}

// ---------------------------------------------------------------------------
// Kernel 3: grouped GEMM1  H = relu(X[gather] @ w1[e] + b1[e]) -> bf16.
// Block = 256 threads (8 waves), C tile 64 (tokens) x 128 (F cols).
// A rows are token-gathered so A is staged manually (TDM gather is <=16 rows).
// ---------------------------------------------------------------------------
__global__ __launch_bounds__(256) void moe_gemm1_kernel(const __bf16* __restrict__ Xbf,
                                                        const float*  __restrict__ w1,
                                                        const float*  __restrict__ b1,
                                                        const int*    __restrict__ cnt,
                                                        const int*    __restrict__ rowStart,
                                                        const int*    __restrict__ tokIdx,
                                                        __bf16* __restrict__ Hbf) {
  const int e     = blockIdx.z;
  const int count = cnt[e];
  const int m0    = blockIdx.y * 64;
  if (m0 >= count) return;
  const int n0 = blockIdx.x * 128;

  __shared__ __align__(16) __bf16 As[64][40];
  __shared__ __align__(16) __bf16 Bs[128][40];
  __shared__ int idxS[64];

  const int tid  = threadIdx.x;
  const int lane = tid & 31;
  const int wave = tid >> 5;
  const int wr   = (wave >> 1) * 16;        // wave's M offset
  const int wc   = (wave & 1) * 4;          // wave's first 16-wide N tile

  if (tid < 64) {
    const int m = m0 + tid;
    idxS[tid] = tokIdx[e * NTOK + ((m < count) ? m : m0)];
  }

  const float* w1e = w1 + (size_t)e * DDIM * FDIM;
  v8f c[4] = {};

  for (int k0 = 0; k0 < DDIM; k0 += 32) {
    __syncthreads();
    {  // stage A: 64x32 gathered bf16 rows, 16B per thread
      const int r  = tid >> 2;
      const int c8 = (tid & 3) * 8;
      *(uint4*)(&As[r][c8]) =
          *(const uint4*)(Xbf + (size_t)idxS[r] * DDIM + k0 + c8);
    }
    stage_b_tile(Bs, w1e, FDIM, k0, n0, tid);
    if (k0 + 32 < DDIM)   // pull next weight tile toward L2 (global_prefetch_b8)
      __builtin_prefetch(w1e + (size_t)(k0 + 32 + (tid >> 3)) * FDIM + n0 + (tid & 7) * 16, 0, 1);
    __syncthreads();

    const v16bf a = load_a_frag(As, lane, wr);
    #pragma unroll
    for (int t = 0; t < 4; ++t) {
      const v16bf b = load_b_frag(Bs, lane, (wc + t) * 16);
      c[t] = __builtin_amdgcn_wmma_f32_16x16x32_bf16(false, a, false, b,
                                                     (short)0, c[t], false, false);
    }
  }

  // Epilogue: bias + relu, bf16 store to H.
  const float* b1e = b1 + (size_t)e * FDIM;
  const int rs = rowStart[e];
  #pragma unroll
  for (int t = 0; t < 4; ++t) {
    const int nc   = n0 + (wc + t) * 16 + (lane & 15);
    const float bv = b1e[nc];
    #pragma unroll
    for (int r = 0; r < 8; ++r) {
      const int mloc = wr + r + 8 * (lane >> 4);
      const int m    = m0 + mloc;
      if (m < count) {
        float v = c[t][r] + bv;
        v = v > 0.f ? v : 0.f;
        Hbf[(size_t)(rs + m) * FDIM + nc] = (__bf16)v;
      }
    }
  }
}

// ---------------------------------------------------------------------------
// Kernel 4: grouped GEMM2  out[tok] += gate * (H @ w2[e] + b2[e]).
// A tile (64x32 bf16 rows of H, row stride FDIM) is moved by the Tensor Data
// Mover directly into the padded [64][40] LDS layout:
//   D# group0: count=1, lds_addr, global_addr, type=2
//   D# group1: data_size=2B, pad 4 DWORDs per 16 DWORDs (-> ld 40),
//              tile 32x64, tensor_dim0=stride=FDIM, tensor_dim1=rows left.
// One wave issues it (EXEC ignored for tensor ops); completion via TENSORcnt.
// ---------------------------------------------------------------------------
__global__ __launch_bounds__(256) void moe_gemm2_kernel(const __bf16* __restrict__ Hbf,
                                                        const float*  __restrict__ w2,
                                                        const float*  __restrict__ b2,
                                                        const int*    __restrict__ cnt,
                                                        const int*    __restrict__ rowStart,
                                                        const int*    __restrict__ tokIdx,
                                                        const float*  __restrict__ gateVal,
                                                        float* __restrict__ out) {
  const int e     = blockIdx.z;
  const int count = cnt[e];
  const int m0    = blockIdx.y * 64;
  if (m0 >= count) return;
  const int n0 = blockIdx.x * 128;

  __shared__ __align__(16) __bf16 As[64][40];
  __shared__ __align__(16) __bf16 Bs[128][40];
  __shared__ int   idxS[64];
  __shared__ float gateS[64];

  const int tid  = threadIdx.x;
  const int lane = tid & 31;
  const int wave = tid >> 5;
  const int wr   = (wave >> 1) * 16;
  const int wc   = (wave & 1) * 4;

  if (tid < 64) {
    const int m  = m0 + tid;
    const int mm = (m < count) ? m : (count - 1);
    idxS[tid]  = tokIdx[e * NTOK + mm];
    gateS[tid] = (m < count) ? gateVal[e * NTOK + mm] : 0.f;
  }

  const float* w2e = w2 + (size_t)e * FDIM * DDIM;
  const int rs = rowStart[e];

  // --- TDM descriptor (uniform) ---
  const unsigned ldsA = (unsigned)(unsigned long long)(&As[0][0]);   // LDS byte offset
  const __bf16*  gp   = Hbf + (size_t)(rs + m0) * FDIM;              // tile base
  const unsigned rows = (unsigned)(count - m0);                      // OOB rows read 0
  u32x8 g1;
  g1[0] = (1u << 16)        // data_size = 2 bytes
        | (1u << 20)        // pad_enable
        | (3u << 22)        // pad_interval: 16 DWORDs (64B) between pads
        | (3u << 25);       // pad_amount: 4 DWORDs (16B) -> row pitch 80B = [40] bf16
  g1[1] = ((unsigned)FDIM & 0xFFFFu) << 16;                     // tensor_dim0 lo16
  g1[2] = (((unsigned)FDIM >> 16) & 0xFFFFu) | ((rows & 0xFFFFu) << 16);
  g1[3] = ((rows >> 16) & 0xFFFFu) | (32u << 16);               // tile_dim0 = 32
  g1[4] = 64u;                                                  // tile_dim1 = 64
  g1[5] = (unsigned)FDIM;                                       // tensor_dim0_stride
  g1[6] = 0u;
  g1[7] = 0u;

  v8f c[4] = {};

  for (int k0 = 0; k0 < FDIM; k0 += 32) {
    __syncthreads();
    if (tid == 0) {  // issue tensor DMA: H tile -> padded LDS A
      const unsigned long long ga = (unsigned long long)(gp + k0);
      u32x4 g0;
      g0[0] = 1u;                                               // count=1 valid D#
      g0[1] = ldsA;
      g0[2] = (unsigned)ga;
      g0[3] = (unsigned)((ga >> 32) & 0x1FFFFFFull) | 0x80000000u;  // addr[56:32] | type=2
      asm volatile("tensor_load_to_lds %0, %1" :: "s"(g0), "s"(g1) : "memory");
    }
    stage_b_tile(Bs, w2e, DDIM, k0, n0, tid);
    if (k0 + 32 < FDIM)
      __builtin_prefetch(w2e + (size_t)(k0 + 32 + (tid >> 3)) * DDIM + n0 + (tid & 7) * 16, 0, 1);
    if (tid == 0) __builtin_amdgcn_s_wait_tensorcnt(0);
    __syncthreads();

    const v16bf a = load_a_frag(As, lane, wr);
    #pragma unroll
    for (int t = 0; t < 4; ++t) {
      const v16bf b = load_b_frag(Bs, lane, (wc + t) * 16);
      c[t] = __builtin_amdgcn_wmma_f32_16x16x32_bf16(false, a, false, b,
                                                     (short)0, c[t], false, false);
    }
  }

  // Epilogue: bias, gate-scale, atomic combine into output.
  const float* b2e = b2 + (size_t)e * DDIM;
  #pragma unroll
  for (int t = 0; t < 4; ++t) {
    const int nc   = n0 + (wc + t) * 16 + (lane & 15);
    const float bv = b2e[nc];
    #pragma unroll
    for (int r = 0; r < 8; ++r) {
      const int mloc = wr + r + 8 * (lane >> 4);
      const int m    = m0 + mloc;
      if (m < count) {
        const float g   = gateS[mloc];
        const int   tok = idxS[mloc];
        atomicAdd(&out[(size_t)tok * DDIM + nc], g * (c[t][r] + bv));
      }
    }
  }
}

// ---------------------------------------------------------------------------
// Host entry.
// ---------------------------------------------------------------------------
extern "C" void kernel_launch(void* const* d_in, const int* in_sizes, int n_in,
                              void* d_out, int out_size, void* d_ws, size_t ws_size,
                              hipStream_t stream) {
  const float* X  = (const float*)d_in[0];
  const float* rw = (const float*)d_in[1];
  const float* rb = (const float*)d_in[2];
  const float* w1 = (const float*)d_in[3];
  const float* b1 = (const float*)d_in[4];
  const float* w2 = (const float*)d_in[5];
  const float* b2 = (const float*)d_in[6];
  float* out = (float*)d_out;

  // Workspace layout (bytes):
  //   [0)      float impSum[8]
  //   [64)     int   cnt[8]
  //   [128)    int   rowStart[8]
  //   [256)    int   tokIdx[E*N]          (256 KB)
  //   [+]      float gateVal[E*N]         (256 KB)
  //   [+]      bf16  Xbf[N*D]             (16 MB)
  //   [+]      bf16  Hbf[N*K rows * 4D]   (128 MB)
  char* ws = (char*)d_ws;
  float*  impSum   = (float*)(ws);
  int*    cnt      = (int*)(ws + 64);
  int*    rowStart = (int*)(ws + 128);
  int*    tokIdx   = (int*)(ws + 256);
  float*  gateVal  = (float*)(ws + 256 + (size_t)NEXP * NTOK * 4);
  __bf16* Xbf      = (__bf16*)(ws + 256 + (size_t)NEXP * NTOK * 8);
  __bf16* Hbf      = (__bf16*)(ws + 256 + (size_t)NEXP * NTOK * 8 +
                               (size_t)NTOK * DDIM * 2);

  // 1) zero accumulators + output
  moe_zero_kernel<<<(NTOK * DDIM / 4) / 256, 256, 0, stream>>>(out, impSum, cnt);
  // 2) router (softmax, top-2, token lists, X -> bf16, importance sums)
  moe_router_kernel<<<NTOK / 8, 256, 0, stream>>>(X, rw, rb, impSum, cnt,
                                                  tokIdx, gateVal, Xbf);
  // 3) prefix offsets + aux loss
  moe_prep_kernel<<<1, 32, 0, stream>>>(impSum, cnt, rowStart,
                                        out + (size_t)NTOK * DDIM);
  // 4) H = relu(X w1 + b1), grouped per expert
  moe_gemm1_kernel<<<dim3(FDIM / 128, NTOK / 64, NEXP), 256, 0, stream>>>(
      Xbf, w1, b1, cnt, rowStart, tokIdx, Hbf);
  // 5) out += gate * (H w2 + b2)
  moe_gemm2_kernel<<<dim3(DDIM / 128, NTOK / 64, NEXP), 256, 0, stream>>>(
      Hbf, w2, b2, cnt, rowStart, tokIdx, gateVal, out);
}
